// MultiCategoricalAutoRegressiveModel_48498770706800
// MI455X (gfx1250) — matrix operand order repaired and checked by reading, not compiled, not run on hardware
//
#include <hip/hip_runtime.h>

typedef __attribute__((ext_vector_type(16))) _Float16 v16h;
typedef __attribute__((ext_vector_type(8)))  _Float16 v8h;
typedef __attribute__((ext_vector_type(8)))  float    v8f;

// ---------------- packed-weight workspace layout (element offsets, f16) ----------------
// Column-permuted / zero-padded so every GEMM input is ONE contiguous LDS view.
// Arena row layout: [ctx(0:256) | sa(256:288) | h(288:416) | seq(416:672)]
#define OFF_WIN   0u                    // 8 x 256 x 288   ([ctx,sa] cols; k=0 zero-padded sa)
#define OFF_WIH   589824u               // 384 x 256
#define OFF_WHH   688128u               // 384 x 128
#define OFF_WS1   737280u               // 128 x 416      ([ctx, 0(sa), h] cols)
#define OFF_WS2   790528u               // 128 x 128
#define OFF_WS3   806912u               // 256 x 128
#define OFF_WOUT  839680u               // 8 x 32 x 672   ([ctx, sa, 0(h), seq] cols)
#define WPK_TOTAL 1011712u              // halves (~2.02 MB)

// ---------------- weight pack kernels (run once per launch; trivial cost) ----------------
__global__ void pack_f16_copy(_Float16* __restrict__ dst, const float* __restrict__ src, int n) {
  int i = blockIdx.x * blockDim.x + threadIdx.x;
  if (i < n) dst[i] = (_Float16)src[i];
}

__global__ void pack_win(_Float16* __restrict__ dst,
                         const float* __restrict__ w_in0, const float* __restrict__ w_in) {
  int i = blockIdx.x * blockDim.x + threadIdx.x;       // 8*256*288
  if (i >= 8 * 256 * 288) return;
  int c = i % 288, r = (i / 288) % 256, k = i / (288 * 256);
  float v;
  if (k == 0) {                                        // w_in0: (256,256), all ctx
    v = (c < 256) ? w_in0[r * 256 + c] : 0.0f;
  } else {                                             // w_in: (256,288) = [sa(32), ctx(256)]
    const float* W = w_in + (size_t)(k - 1) * 256 * 288;
    v = (c < 256) ? W[r * 288 + 32 + c] : W[r * 288 + (c - 256)];
  }
  dst[i] = (_Float16)v;
}

__global__ void pack_ws1(_Float16* __restrict__ dst, const float* __restrict__ w_s1) {
  int i = blockIdx.x * blockDim.x + threadIdx.x;       // 128*416
  if (i >= 128 * 416) return;
  int c = i % 416, r = i / 416;                        // w_s1: (128,384) = [h(128), ctx(256)]
  float v = (c < 256) ? w_s1[r * 384 + 128 + c]
          : (c < 288) ? 0.0f
                      : w_s1[r * 384 + (c - 288)];
  dst[i] = (_Float16)v;
}

__global__ void pack_wout(_Float16* __restrict__ dst,
                          const float* __restrict__ w_out0, const float* __restrict__ w_out) {
  int i = blockIdx.x * blockDim.x + threadIdx.x;       // 8*32*672
  if (i >= 8 * 32 * 672) return;
  int c = i % 672, r = (i / 672) % 32, k = i / (672 * 32);
  float v;
  if (k == 0) {                                        // w_out0: (32,512) = [seq(256), ctx(256)]
    if (c < 256)      v = w_out0[r * 512 + 256 + c];   // ctx
    else if (c < 416) v = 0.0f;                        // sa + h pad
    else              v = w_out0[r * 512 + (c - 416)]; // seq
  } else {                                             // w_out: (32,544) = [seq(256), sa(32), ctx(256)]
    const float* W = w_out + (size_t)(k - 1) * 32 * 544;
    if (c < 256)      v = W[r * 544 + 288 + c];        // ctx
    else if (c < 288) v = W[r * 544 + c];              // sa
    else if (c < 416) v = 0.0f;                        // h pad
    else              v = W[r * 544 + (c - 416)];      // seq
  }
  dst[i] = (_Float16)v;
}

// ---------------- fragment helpers ----------------
static __device__ inline v8f vzero8() {
  v8f r;
#pragma unroll
  for (int i = 0; i < 8; ++i) r[i] = 0.0f;
  return r;
}

// A-fragment (16x32 f16, MxK) from LDS, row-major.
// lanes 0-15: m=lane, K=0..7 & 16..23 ; lanes 16-31: m=lane-16, K=8..15 & 24..31.
static __device__ inline v16h load_a_lds(const _Float16* base, int lda, int koff, int lane) {
  const int m  = lane & 15;
  const int kb = (lane < 16) ? 0 : 8;
  const _Float16* p = base + m * lda + koff + kb;
  v8h lo = *(const v8h*)(p);
  v8h hi = *(const v8h*)(p + 16);
  v16h r;
#pragma unroll
  for (int i = 0; i < 8; ++i) { r[i] = lo[i]; r[i + 8] = hi[i]; }
  return r;
}

// B-fragment (32x16 f16, KxN) from packed f16 global, (out,in) row-major.
// lanes 0-15: n=lane, K=0..7 & 16..23 ; lanes 16-31: n=lane-16, K=8..15 & 24..31.
static __device__ inline v16h load_b_w(const _Float16* __restrict__ W, int ldw,
                                       int n0, int kc, int lane) {
  const int n  = lane & 15;
  const int ks = (lane < 16) ? 0 : 8;
  const _Float16* p = W + (size_t)(n0 + n) * ldw + kc + ks;
  v8h lo = *(const v8h*)(p);
  v8h hi = *(const v8h*)(p + 16);
  v16h r;
#pragma unroll
  for (int i = 0; i < 8; ++i) { r[i] = lo[i]; r[i + 8] = hi[i]; }
  return r;
}

// Double-M-tile GEMM: both 16-row A tiles share every B fragment.
// acc0 += A[0:16)xK * W^T tile ; acc1 += A[16:32)xK * W^T tile
static __device__ inline void gemm_acc2(const _Float16* A, int lda, int Kdim,
                                        const _Float16* __restrict__ W, int ldw, int n0,
                                        int lane, v8f& acc0, v8f& acc1) {
#pragma unroll 2
  for (int kc = 0; kc < Kdim; kc += 32) {
    v16h a0 = load_a_lds(A, lda, kc, lane);
    v16h a1 = load_a_lds(A + 16 * lda, lda, kc, lane);
    v16h b  = load_b_w(W, ldw, n0, kc, lane);
    acc0 = __builtin_amdgcn_wmma_f32_16x16x32_f16(false, a0, false, b, (short)0, acc0, false, false);
    acc1 = __builtin_amdgcn_wmma_f32_16x16x32_f16(false, a1, false, b, (short)0, acc1, false, false);
  }
}

// Scatter 16x16 f32 D-tile into LDS as f16, row-major.
static __device__ inline void store_d_lds(_Float16* base, int ldd, int n0, int lane, v8f acc) {
  const int n  = (lane & 15) + n0;
  const int mb = (lane < 16) ? 0 : 8;
#pragma unroll
  for (int g = 0; g < 8; ++g)
    base[(mb + g) * ldd + n] = (_Float16)acc[g];
}

static __device__ inline float sigmoidf_(float v) { return 1.0f / (1.0f + __expf(-v)); }

#define NW 2                 // waves per block
#define MROWS 32             // batch rows per wave (2 x 16-row WMMA M-tiles)
#define ARENA_LD 672         // arena row stride (halves)
// per-wave LDS (halves): arena 32x672 + rnn 32x256 + s 32x128 + s2 32x128 + hp0/hp1 32x128
#define WV_HALVES (21504 + 8192 + 4096 + 4096 + 4096 + 4096)   // 46080 (92,160 B)

__global__ __launch_bounds__(32 * NW, 1)
void marm_kernel(const float* __restrict__ x,    const float* __restrict__ ctx,
                 const _Float16* __restrict__ wpk,
                 const float* __restrict__ b_in0, const float* __restrict__ b_in,
                 const float* __restrict__ bih,   const float* __restrict__ bhh,
                 const float* __restrict__ b_s1,  const float* __restrict__ b_s2,
                 const float* __restrict__ b_s3,
                 const float* __restrict__ b_out0,const float* __restrict__ b_out,
                 float* __restrict__ out)
{
  extern __shared__ __align__(16) _Float16 smem[];
  const int lane = threadIdx.x & 31;
  const int wave = threadIdx.x >> 5;
  const int row0 = (blockIdx.x * NW + wave) * MROWS;

  _Float16* sm    = smem + wave * WV_HALVES;
  _Float16* arena = sm;                    // 32 x 672: [ctx | sa | h | seq]
  _Float16* rnnB  = arena + 32 * ARENA_LD; // 32 x 256 rnn_in
  _Float16* sB    = rnnB + 32 * 256;       // 32 x 128 s1-out
  _Float16* s2B   = sB   + 32 * 128;       // 32 x 128 s2-out
  _Float16* hp0   = s2B  + 32 * 128;       // 32 x 128 hidden (prev), double-buffered
  _Float16* hp1   = hp0  + 32 * 128;

  const _Float16* Wih = wpk + OFF_WIH;
  const _Float16* Whh = wpk + OFF_WHH;
  const _Float16* Ws1 = wpk + OFF_WS1;
  const _Float16* Ws2 = wpk + OFF_WS2;
  const _Float16* Ws3 = wpk + OFF_WS3;

  // setup: stage ctx (fp32->f16) into arena ctx slot, zero sa slot + hidden state
#pragma unroll 1
  for (int i = lane; i < 32 * 256; i += 32)
    arena[(i >> 8) * ARENA_LD + (i & 255)] = (_Float16)ctx[(size_t)(row0 + (i >> 8)) * 256 + (i & 255)];
#pragma unroll 1
  for (int i = lane; i < 32 * 32; i += 32)
    arena[(i >> 5) * ARENA_LD + 256 + (i & 31)] = (_Float16)0.0f;
#pragma unroll 1
  for (int i = lane; i < 32 * 128; i += 32) hp0[i] = (_Float16)0.0f;

  _Float16* hprev = hp0;
  _Float16* hnext = hp1;

  const int n  = lane & 15;
  const int mb = (lane < 16) ? 0 : 8;

#pragma unroll 1
  for (int k = 0; k < 8; ++k) {
    const _Float16* Wi = wpk + OFF_WIN + (size_t)k * 256 * 288;
    const _Float16* Wo = wpk + OFF_WOUT + (size_t)k * 32 * 672;
    const float* Bi = (k == 0) ? b_in0 : b_in + (size_t)(k - 1) * 256;
    const float* Bo = (k == 0) ? b_out0 : b_out + (size_t)(k - 1) * 32;
    if (k < 7) {  // warm L2 for next step's weights
      __builtin_prefetch(Wi + 256 * 288, 0, 1);
      __builtin_prefetch(Wo + 32 * 672, 0, 1);
    }

    // ---- rnn_in = [ctx,sa] @ Wi'^T + bi  (K=288, arena view) ----
#pragma unroll 1
    for (int nt = 0; nt < 256; nt += 16) {
      v8f a0 = vzero8(), a1 = vzero8();
      gemm_acc2(arena, ARENA_LD, 288, Wi, 288, nt, lane, a0, a1);
      const float bv = Bi[nt + n];
#pragma unroll
      for (int g = 0; g < 8; ++g) { a0[g] += bv; a1[g] += bv; }
      store_d_lds(rnnB,            256, nt, lane, a0);
      store_d_lds(rnnB + 16 * 256, 256, nt, lane, a1);
    }

    // ---- GRU cell: 6 double-tile GEMMs per 16-col slice, gates fused in f32 regs ----
#pragma unroll 1
    for (int jt = 0; jt < 128; jt += 16) {
      v8f ir[2], iz[2], inn[2], hr[2], hz[2], hn[2];
#pragma unroll
      for (int rt = 0; rt < 2; ++rt) {
        ir[rt] = vzero8(); iz[rt] = vzero8(); inn[rt] = vzero8();
        hr[rt] = vzero8(); hz[rt] = vzero8(); hn[rt] = vzero8();
      }
      gemm_acc2(rnnB, 256, 256, Wih, 256, jt,       lane, ir[0],  ir[1]);
      gemm_acc2(rnnB, 256, 256, Wih, 256, 128 + jt, lane, iz[0],  iz[1]);
      gemm_acc2(rnnB, 256, 256, Wih, 256, 256 + jt, lane, inn[0], inn[1]);
      gemm_acc2(hprev, 128, 128, Whh, 128, jt,       lane, hr[0], hr[1]);
      gemm_acc2(hprev, 128, 128, Whh, 128, 128 + jt, lane, hz[0], hz[1]);
      gemm_acc2(hprev, 128, 128, Whh, 128, 256 + jt, lane, hn[0], hn[1]);
      const float bir = bih[jt + n], biz = bih[128 + jt + n], binn = bih[256 + jt + n];
      const float bhr = bhh[jt + n], bhz = bhh[128 + jt + n], bhn  = bhh[256 + jt + n];
#pragma unroll
      for (int rt = 0; rt < 2; ++rt) {
#pragma unroll
        for (int g = 0; g < 8; ++g) {
          const int rr = 16 * rt + mb + g;
          const float r  = sigmoidf_(ir[rt][g] + bir + hr[rt][g] + bhr);
          const float z  = sigmoidf_(iz[rt][g] + biz + hz[rt][g] + bhz);
          const float nn = tanhf(inn[rt][g] + binn + r * (hn[rt][g] + bhn));
          const float ho = (float)hprev[rr * 128 + jt + n];
          const float hv = (1.0f - z) * nn + z * ho;
          arena[rr * ARENA_LD + 288 + jt + n] = (_Float16)hv;  // h slot (s1 / out views)
          hnext[rr * 128 + jt + n] = (_Float16)hv;             // next step's gh input
        }
      }
    }

    // ---- s = relu([ctx, 0, h] @ Ws1'^T + b_s1)  (K=416, arena view) ----
#pragma unroll 1
    for (int nt = 0; nt < 128; nt += 16) {
      v8f a0 = vzero8(), a1 = vzero8();
      gemm_acc2(arena, ARENA_LD, 416, Ws1, 416, nt, lane, a0, a1);
      const float bv = b_s1[nt + n];
#pragma unroll
      for (int g = 0; g < 8; ++g) { a0[g] = fmaxf(a0[g] + bv, 0.0f); a1[g] = fmaxf(a1[g] + bv, 0.0f); }
      store_d_lds(sB,            128, nt, lane, a0);
      store_d_lds(sB + 16 * 128, 128, nt, lane, a1);
    }

    // ---- s = relu(s @ Ws2^T + b_s2) ----
#pragma unroll 1
    for (int nt = 0; nt < 128; nt += 16) {
      v8f a0 = vzero8(), a1 = vzero8();
      gemm_acc2(sB, 128, 128, Ws2, 128, nt, lane, a0, a1);
      const float bv = b_s2[nt + n];
#pragma unroll
      for (int g = 0; g < 8; ++g) { a0[g] = fmaxf(a0[g] + bv, 0.0f); a1[g] = fmaxf(a1[g] + bv, 0.0f); }
      store_d_lds(s2B,            128, nt, lane, a0);
      store_d_lds(s2B + 16 * 128, 128, nt, lane, a1);
    }

    // ---- seq_out = s @ Ws3^T + b_s3 -> arena seq slot ----
#pragma unroll 1
    for (int nt = 0; nt < 256; nt += 16) {
      v8f a0 = vzero8(), a1 = vzero8();
      gemm_acc2(s2B, 128, 128, Ws3, 128, nt, lane, a0, a1);
      const float bv = b_s3[nt + n];
#pragma unroll
      for (int g = 0; g < 8; ++g) { a0[g] += bv; a1[g] += bv; }
      store_d_lds(arena,                 ARENA_LD, 416 + nt, lane, a0);
      store_d_lds(arena + 16 * ARENA_LD, ARENA_LD, 416 + nt, lane, a1);
    }

    // ---- logits = [ctx, sa, 0, seq] @ Wo'^T + bo  (K=672, N=32) ----
    v8f accA[2] = { vzero8(), vzero8() };
    v8f accB[2] = { vzero8(), vzero8() };
    gemm_acc2(arena, ARENA_LD, 672, Wo, 672, 0,  lane, accA[0], accA[1]);
    gemm_acc2(arena, ARENA_LD, 672, Wo, 672, 16, lane, accB[0], accB[1]);
    const float bo0 = Bo[n], bo1 = Bo[16 + n];

    // one-hot targets + log-softmax + lp, per 16-row block
#pragma unroll
    for (int rt = 0; rt < 2; ++rt) {
      float saLo[8], saHi[8];
#pragma unroll
      for (int g = 0; g < 8; ++g) {
        const size_t row = (size_t)(row0 + 16 * rt + mb + g) * 256 + (size_t)k * 32;
        saLo[g] = x[row + n];
        saHi[g] = x[row + 16 + n];
      }
#pragma unroll
      for (int g = 0; g < 8; ++g) {
        const float la = accA[rt][g] + bo0;
        const float lb = accB[rt][g] + bo1;
        float mx = fmaxf(la, lb);
#pragma unroll
        for (int d = 8; d >= 1; d >>= 1) mx = fmaxf(mx, __shfl_xor(mx, d, 32));
        float se = __expf(la - mx) + __expf(lb - mx);
#pragma unroll
        for (int d = 8; d >= 1; d >>= 1) se += __shfl_xor(se, d, 32);
        const float lse = mx + __logf(se);
        float lp = saLo[g] * (la - lse) + saHi[g] * (lb - lse);
#pragma unroll
        for (int d = 8; d >= 1; d >>= 1) lp += __shfl_xor(lp, d, 32);
        if (n == 0) out[(size_t)(row0 + 16 * rt + mb + g) * 8 + k] = lp;
      }
      // stash one-hot into arena sa slot (next step's reg / out views)
#pragma unroll
      for (int g = 0; g < 8; ++g) {
        const int rr = 16 * rt + mb + g;
        arena[rr * ARENA_LD + 256 + n]      = (_Float16)saLo[g];
        arena[rr * ARENA_LD + 256 + 16 + n] = (_Float16)saHi[g];
      }
    }

    _Float16* t = hprev; hprev = hnext; hnext = t;  // swap hidden buffers
  }
}

extern "C" void kernel_launch(void* const* d_in, const int* in_sizes, int n_in,
                              void* d_out, int out_size, void* d_ws, size_t ws_size,
                              hipStream_t stream) {
  (void)in_sizes; (void)n_in; (void)out_size; (void)ws_size;
  const float* x      = (const float*)d_in[0];
  const float* ctx    = (const float*)d_in[1];
  const float* w_in0  = (const float*)d_in[2];
  const float* b_in0  = (const float*)d_in[3];
  const float* w_in   = (const float*)d_in[4];
  const float* b_in   = (const float*)d_in[5];
  const float* wih    = (const float*)d_in[6];
  const float* whh    = (const float*)d_in[7];
  const float* bih    = (const float*)d_in[8];
  const float* bhh    = (const float*)d_in[9];
  const float* w_s1   = (const float*)d_in[10];
  const float* b_s1   = (const float*)d_in[11];
  const float* w_s2   = (const float*)d_in[12];
  const float* b_s2   = (const float*)d_in[13];
  const float* w_s3   = (const float*)d_in[14];
  const float* b_s3   = (const float*)d_in[15];
  const float* w_out0 = (const float*)d_in[16];
  const float* b_out0 = (const float*)d_in[17];
  const float* w_out  = (const float*)d_in[18];
  const float* b_out  = (const float*)d_in[19];
  float* out = (float*)d_out;
  _Float16* wpk = (_Float16*)d_ws;

  // ---- pack weights to f16 (column-permuted / zero-padded) into d_ws ----
  pack_win <<<(8 * 256 * 288 + 255) / 256, 256, 0, stream>>>(wpk + OFF_WIN, w_in0, w_in);
  pack_f16_copy<<<(384 * 256 + 255) / 256, 256, 0, stream>>>(wpk + OFF_WIH, wih, 384 * 256);
  pack_f16_copy<<<(384 * 128 + 255) / 256, 256, 0, stream>>>(wpk + OFF_WHH, whh, 384 * 128);
  pack_ws1 <<<(128 * 416 + 255) / 256, 256, 0, stream>>>(wpk + OFF_WS1, w_s1);
  pack_f16_copy<<<(128 * 128 + 255) / 256, 256, 0, stream>>>(wpk + OFF_WS2, w_s2, 128 * 128);
  pack_f16_copy<<<(256 * 128 + 255) / 256, 256, 0, stream>>>(wpk + OFF_WS3, w_s3, 256 * 128);
  pack_wout<<<(8 * 32 * 672 + 255) / 256, 256, 0, stream>>>(wpk + OFF_WOUT, w_out0, w_out);

  // ---- fused autoregressive model ----
  const size_t shmem = (size_t)NW * WV_HALVES * sizeof(_Float16);  // 184,320 B (< 320 KB WGP LDS)
  (void)hipFuncSetAttribute(reinterpret_cast<const void*>(marm_kernel),
                            hipFuncAttributeMaxDynamicSharedMemorySize, (int)shmem);
  dim3 grid(65536 / (MROWS * NW));  // 1024 blocks
  dim3 block(32 * NW);              // 2 waves
  marm_kernel<<<grid, block, shmem, stream>>>(x, ctx, wpk,
                                              b_in0, b_in, bih, bhh,
                                              b_s1, b_s2, b_s3, b_out0, b_out, out);
}
